// MultiheadFlashlinearrope_24635932410426
// MI455X (gfx1250) — matrix operand order repaired and checked by reading, not compiled, not run on hardware
//
#include <hip/hip_runtime.h>

// ---------------------------------------------------------------------------
// MultiheadFlashlinearrope for MI455X (gfx1250, wave32, WMMA bf16)
//
// ~584 GFLOP total (4x [16384x2048x2048] GEMMs + 8192 chunk-local 128^3
// matmuls) vs a few hundred MB of traffic @23.3 TB/s -> matrix-throughput
// bound. All matmuls go through v_wmma_f32_16x16x32_bf16. All LDS fragment
// reads are contiguous ds_load_b128 (A row-major, B staged transposed), all
// LDS stages are vector copies, and all decay factors are exp2f(L*x) with a
// hoisted per-head L = log2(r) (one FMA + one transcendental each).
// ---------------------------------------------------------------------------

namespace {
constexpr int Bz  = 4;
constexpr int Tz  = 4096;
constexpr int Hz  = 16;
constexpr int Dz  = 128;
constexpr int Ez  = 2048;
constexpr int Cz  = 128;
constexpr int NCz = 32;
constexpr int Mz  = Bz * Tz;   // 16384
constexpr int LDC = 136;       // chunk-tile LDS stride (mult of 8, 272B rows)
constexpr float LOG2E = 1.4426950408889634f;
}

typedef __attribute__((ext_vector_type(16))) __bf16 v16bf;
typedef __attribute__((ext_vector_type(8)))  __bf16 v8bf;
typedef __attribute__((ext_vector_type(8)))  float  v8f;

// log2(r) for head h: r = exp(-slope), slope = 2^(-0.5*(h+1))
__device__ __forceinline__ float head_log2r(int h) {
  return -exp2f(-0.5f * (float)(h + 1)) * LOG2E;
}

// ---------------- WMMA helpers (CDNA5 16x16x32 bf16, wave32) ----------------

__device__ __forceinline__ v8f wmma_bf16(v16bf a, v16bf b, v8f c) {
  return __builtin_amdgcn_wmma_f32_16x16x32_bf16(
      false, a, false, b, (short)0, c, false, false);
}

// A fragment (16x32 MxK): lane 0-15 -> M=lane, K=0..15 (lane 16-31:
// K=16..31). Contiguous -> two b128 LDS loads.
__device__ __forceinline__ v16bf load_a_frag(const __bf16* A, int lda,
                                             int m0, int k0, int lane) {
  const __bf16* p = A + (size_t)(m0 + (lane & 15)) * lda + k0 + ((lane >> 4) << 4);
  const v8bf lo = *(const v8bf*)p;
  const v8bf hi = *(const v8bf*)(p + 8);
  return __builtin_shufflevector(lo, hi, 0, 1, 2, 3, 4, 5, 6, 7,
                                 8, 9, 10, 11, 12, 13, 14, 15);
}

// B fragment (32x16 KxN) from TRANSPOSED tile BT[n][k]: lane 0-15 -> N=lane,
// K=0..15 (lane 16-31: K=16..31). Contiguous -> two b128 LDS loads.
__device__ __forceinline__ v16bf load_bT_frag(const __bf16* BT, int ldt,
                                              int n0, int k0, int lane) {
  const __bf16* p = BT + (size_t)(n0 + (lane & 15)) * ldt + k0 + ((lane >> 4) << 4);
  const v8bf lo = *(const v8bf*)p;
  const v8bf hi = *(const v8bf*)(p + 8);
  return __builtin_shufflevector(lo, hi, 0, 1, 2, 3, 4, 5, 6, 7,
                                 8, 9, 10, 11, 12, 13, 14, 15);
}

__device__ __forceinline__ void zero_acc(v8f acc[2][4]) {
  v8f z = {0.f, 0.f, 0.f, 0.f, 0.f, 0.f, 0.f, 0.f};
#pragma unroll
  for (int i = 0; i < 2; ++i)
#pragma unroll
    for (int j = 0; j < 4; ++j) acc[i][j] = z;
}

// 128x128x128 matmul, A row-major + BT transposed, both LDS-resident.
__device__ __forceinline__ void mm128(const __bf16* A, const __bf16* BT, int ld,
                                      int wm, int wn, int lane, v8f acc[2][4]) {
#pragma unroll
  for (int kk = 0; kk < 128; kk += 32) {
    v16bf af[2], bfr[4];
#pragma unroll
    for (int i = 0; i < 2; ++i) af[i] = load_a_frag(A, ld, wm + i * 16, kk, lane);
#pragma unroll
    for (int j = 0; j < 4; ++j) bfr[j] = load_bT_frag(BT, ld, wn + j * 16, kk, lane);
#pragma unroll
    for (int i = 0; i < 2; ++i)
#pragma unroll
      for (int j = 0; j < 4; ++j) acc[i][j] = wmma_bf16(af[i], bfr[j], acc[i][j]);
  }
}

// ---------------------------- fp32 -> bf16 convert ---------------------------

__global__ __launch_bounds__(256) void cvt_f32_bf16(const float* __restrict__ in,
                                                    __bf16* __restrict__ out, int n4) {
  int i = blockIdx.x * 256 + threadIdx.x;
  if (i >= n4) return;
  const float4 v = reinterpret_cast<const float4*>(in)[i];
  __bf16* o = out + (size_t)i * 4;
  o[0] = (__bf16)v.x; o[1] = (__bf16)v.y; o[2] = (__bf16)v.z; o[3] = (__bf16)v.w;
}

// --------------- weight convert + transpose: f32 [K,N] -> bf16 [N,K] ---------

__global__ __launch_bounds__(256) void cvt_transpose_w(const float* __restrict__ in,
                                                       __bf16* __restrict__ out) {
  __shared__ float tile[32][33];
  const int k0 = blockIdx.y * 32, n0 = blockIdx.x * 32;
  const int tx = threadIdx.x & 31, ty = threadIdx.x >> 5;   // 8 rows per pass
#pragma unroll
  for (int i = 0; i < 32; i += 8)
    tile[ty + i][tx] = in[(size_t)(k0 + ty + i) * Ez + n0 + tx];
  __syncthreads();
#pragma unroll
  for (int i = 0; i < 32; i += 8)
    out[(size_t)(n0 + ty + i) * Ez + k0 + tx] = (__bf16)tile[tx][ty + i];
}

// ------- bf16 transpose per (b,h): [T,D] -> [D,T] (for k after RoPE) ---------

__global__ __launch_bounds__(256) void transpose_dt(const __bf16* __restrict__ in,
                                                    __bf16* __restrict__ out) {
  __shared__ __bf16 tile[32][40];
  const int bh = blockIdx.z;
  const int t0 = blockIdx.x * 32, d0 = blockIdx.y * 32;
  const int tx = threadIdx.x & 31, ty = threadIdx.x >> 5;
  const __bf16* src = in + ((size_t)bh * Tz + t0) * Dz + d0;
  __bf16* dst = out + ((size_t)bh * Dz + d0) * Tz + t0;
#pragma unroll
  for (int i = 0; i < 32; i += 8) tile[ty + i][tx] = src[(size_t)(ty + i) * Dz + tx];
  __syncthreads();
#pragma unroll
  for (int i = 0; i < 32; i += 8) dst[(size_t)(ty + i) * Tz + tx] = tile[tx][ty + i];
}

// ------------------------------- WMMA GEMM ----------------------------------
// C[M,N] = A[M,K] @ WT[N,K]^T. 128x128 block tile, K-tile 64, 8 waves.
// OUTMODE 0: f32 row-major [M,N]
// OUTMODE 1: bf16 scatter to [B,H,T,D]   (q, k)
// OUTMODE 2: bf16 packed v8bf stores to [B,H,D,T]   (v, pre-transposed)

template <int OUTMODE>
__global__ __launch_bounds__(256) void gemm_bf16_128(const __bf16* __restrict__ A,
                                                     const __bf16* __restrict__ WT,
                                                     void* __restrict__ Cout) {
  constexpr int K = Ez, N = Ez;
  __shared__ __bf16 As[128][72];    // 144B rows, 16B aligned
  __shared__ __bf16 BTs[128][72];   // BT[n][k]
  const int tid  = threadIdx.x;
  const int lane = tid & 31;
  const int wave = tid >> 5;
  const int bm = blockIdx.y * 128;
  const int bn = blockIdx.x * 128;
  const int wm = (wave >> 1) * 32;
  const int wn = (wave & 1) * 64;

  v8f acc[2][4];
  zero_acc(acc);

  for (int kk = 0; kk < K; kk += 64) {
#pragma unroll
    for (int q = 0; q < 4; ++q) {       // A tile 128x64: 1024 x 16B vectors
      const int v = tid * 4 + q;
      const int r = v >> 3, c = (v & 7) << 3;
      *reinterpret_cast<uint4*>(&As[r][c]) =
          *reinterpret_cast<const uint4*>(A + (size_t)(bm + r) * K + kk + c);
    }
#pragma unroll
    for (int q = 0; q < 4; ++q) {       // BT tile 128x64 from WT[N,K]
      const int v = tid * 4 + q;
      const int r = v >> 3, c = (v & 7) << 3;
      *reinterpret_cast<uint4*>(&BTs[r][c]) =
          *reinterpret_cast<const uint4*>(WT + (size_t)(bn + r) * K + kk + c);
    }
    __syncthreads();
#pragma unroll
    for (int k0 = 0; k0 < 64; k0 += 32) {
      v16bf af[2], bfr[4];
#pragma unroll
      for (int i = 0; i < 2; ++i) af[i] = load_a_frag(&As[0][0], 72, wm + i * 16, k0, lane);
#pragma unroll
      for (int j = 0; j < 4; ++j) bfr[j] = load_bT_frag(&BTs[0][0], 72, wn + j * 16, k0, lane);
#pragma unroll
      for (int i = 0; i < 2; ++i)
#pragma unroll
        for (int j = 0; j < 4; ++j) acc[i][j] = wmma_bf16(af[i], bfr[j], acc[i][j]);
    }
    __syncthreads();
  }

  const int hi = lane >> 4, nn = lane & 15;
  if (OUTMODE == 0) {
    float* C = (float*)Cout;
#pragma unroll
    for (int i = 0; i < 2; ++i)
#pragma unroll
      for (int j = 0; j < 4; ++j) {
        const int n = bn + wn + j * 16 + nn;
#pragma unroll
        for (int rg = 0; rg < 8; ++rg) {
          const int m = bm + wm + i * 16 + rg + 8 * hi;
          C[(size_t)m * N + n] = acc[i][j][rg];
        }
      }
  } else if (OUTMODE == 1) {
    __bf16* C = (__bf16*)Cout;          // [B,H,T,D]
#pragma unroll
    for (int i = 0; i < 2; ++i)
#pragma unroll
      for (int j = 0; j < 4; ++j) {
        const int n = bn + wn + j * 16 + nn;
        const int h = n >> 7, dd = n & 127;
#pragma unroll
        for (int rg = 0; rg < 8; ++rg) {
          const int m = bm + wm + i * 16 + rg + 8 * hi;
          const int bq = m >> 12, t = m & (Tz - 1);
          C[(((size_t)bq * Hz + h) * Tz + t) * Dz + dd] = (__bf16)acc[i][j][rg];
        }
      }
  } else {
    __bf16* C = (__bf16*)Cout;          // [B,H,D,T], 8 consecutive t per lane
#pragma unroll
    for (int i = 0; i < 2; ++i)
#pragma unroll
      for (int j = 0; j < 4; ++j) {
        const int n = bn + wn + j * 16 + nn;
        const int h = n >> 7, dd = n & 127;
        const int mb = bm + wm + i * 16 + 8 * hi;
        const int bq = mb >> 12, t = mb & (Tz - 1);
        v8bf pk;
#pragma unroll
        for (int rg = 0; rg < 8; ++rg) pk[rg] = (__bf16)acc[i][j][rg];
        *reinterpret_cast<v8bf*>(C + (((size_t)bq * Hz + h) * Dz + dd) * Tz + t) = pk;
      }
  }
}

// --------------------------- RoPE (in place, bf16) ---------------------------
// D/4=32 nonzero freqs -> pairs (j, j+64); j in [32,64) are identity.
// freq_j = (1/1024)^(j/31) = exp2(-10*j/31)

__global__ __launch_bounds__(256) void rope_kernel(__bf16* __restrict__ q) {
  const size_t p = (size_t)blockIdx.x * 256 + threadIdx.x;   // (b,h,t,j)
  const int j = (int)(p & 63);
  const size_t row = p >> 6;                                 // (b*H+h)*T + t
  if (j >= 32) return;
  const int t = (int)(row & (Tz - 1));
  __bf16* base = q + row * Dz;
  const float f = exp2f(-10.0f * (float)j * (1.0f / 31.0f));
  float s, c;
  __sincosf((float)t * f, &s, &c);
  const float x1 = (float)base[j], x2 = (float)base[j + 64];
  base[j]      = (__bf16)(x1 * c + x2 * s);
  base[j + 64] = (__bf16)(-x1 * s + x2 * c);
}

// ---------------- per-chunk decayed KV outer product (WMMA) ------------------
// KVT[e][d] = KV[d][e] = sum_s r^(C-1-s) * k[s][d] * v[s][e]
// A[d][s] = ktT straight; BT[e][s] = vtT straight * kdec(s); out packed v8bf.

__global__ __launch_bounds__(256) void chunk_kv_kernel(const __bf16* __restrict__ ktT,
                                                       const __bf16* __restrict__ vtT,
                                                       __bf16* __restrict__ kvcT) {
  __shared__ __bf16 As[128][LDC];
  __shared__ __bf16 BTs[128][LDC];
  const int nc = blockIdx.x, h = blockIdx.y, b = blockIdx.z;
  const int tid = threadIdx.x, lane = tid & 31, wave = tid >> 5;
  const int wm = (wave >> 1) * 32, wn = (wave & 1) * 64;
  const float L = head_log2r(h);            // log2(r), r^x = exp2f(L*x)
  const size_t bh = (size_t)b * Hz + h;
  const int rr = tid >> 1, c0 = (tid & 1) * 64;
  const int t0 = nc * Cz;
  const __bf16* ksrc = ktT + ((size_t)bh * Dz + rr) * Tz + t0 + c0;
  const __bf16* vsrc = vtT + ((size_t)bh * Dz + rr) * Tz + t0 + c0;
#pragma unroll
  for (int j = 0; j < 64; j += 8) {
    *reinterpret_cast<uint4*>(&As[rr][c0 + j]) =
        *reinterpret_cast<const uint4*>(ksrc + j);
    const v8bf x = *reinterpret_cast<const v8bf*>(vsrc + j);
    v8bf y;
#pragma unroll
    for (int u = 0; u < 8; ++u)
      y[u] = (__bf16)((float)x[u] * exp2f(L * (float)(Cz - 1 - (c0 + j + u))));
    *reinterpret_cast<v8bf*>(&BTs[rr][c0 + j]) = y;
  }
  __syncthreads();
  v8f acc[2][4];
  zero_acc(acc);
  mm128(&As[0][0], &BTs[0][0], LDC, wm, wn, lane, acc);

  __bf16* dst = kvcT + (bh * NCz + nc) * (size_t)(Dz * Dz);   // [e][d]
  const int hi = lane >> 4, nn = lane & 15;
#pragma unroll
  for (int i = 0; i < 2; ++i)
#pragma unroll
    for (int j = 0; j < 4; ++j) {
      const int e = wn + j * 16 + nn;
      const int db = wm + i * 16 + 8 * hi;                    // 8 consecutive d
      v8bf pk;
#pragma unroll
      for (int rg = 0; rg < 8; ++rg) pk[rg] = (__bf16)acc[i][j][rg];
      *reinterpret_cast<v8bf*>(dst + (size_t)e * Dz + db) = pk;
    }
}

// ---------------- exponential prefix scan over chunks (element-parallel) -----

__global__ __launch_bounds__(256) void kv_scan_kernel(const __bf16* __restrict__ kvcT,
                                                      __bf16* __restrict__ kvinT) {
  const size_t idx = (size_t)blockIdx.x * 256 + threadIdx.x;  // [0, B*H*D*D)
  const size_t bh = idx >> 14;
  const size_t de = idx & 16383;
  const int h = (int)(bh & (Hz - 1));
  const float rC = exp2f(head_log2r(h) * (float)Cz);
  const __bf16* src = kvcT  + bh * (size_t)(NCz * 16384) + de;
  __bf16*       dst = kvinT + bh * (size_t)(NCz * 16384) + de;
  float s = 0.f;
#pragma unroll
  for (int nc = 0; nc < NCz; ++nc) {
    dst[(size_t)nc * 16384] = (__bf16)s;
    s = rC * s + (float)src[(size_t)nc * 16384];
  }
}

// ---------------- per-chunk output: scores^T, inter, intra -------------------
// All stages are vector copies; scores computed transposed so the dmask write
// into the A-layout tile is a packed v8bf store.

__global__ __launch_bounds__(256) void chunk_out_kernel(const __bf16* __restrict__ qt,
                                                        const __bf16* __restrict__ kt,
                                                        const __bf16* __restrict__ vtT,
                                                        const __bf16* __restrict__ kvinT,
                                                        __bf16* __restrict__ attnb) {
  __shared__ __bf16 buf1[128][LDC];   // Q (A + BT roles), later V^T
  __shared__ __bf16 buf2[128][LDC];   // K, then KV^T, then scores
  const int nc = blockIdx.x, h = blockIdx.y, b = blockIdx.z;
  const int tid = threadIdx.x, lane = tid & 31, wave = tid >> 5;
  const int wm = (wave >> 1) * 32, wn = (wave & 1) * 64;
  const float L = head_log2r(h);            // log2(r)
  const size_t bh = (size_t)b * Hz + h;
  const int rr = tid >> 1, c0 = (tid & 1) * 64;
  const int t0 = nc * Cz;
  const int hi = lane >> 4, nn = lane & 15;

  // phase 1: buf1 = Q[c][d], buf2 = K[s][d]  (both straight [T,D] copies)
  {
    const size_t rowoff = (bh * Tz + (size_t)t0 + rr) * Dz + c0;
#pragma unroll
    for (int j = 0; j < 64; j += 8) {
      *reinterpret_cast<uint4*>(&buf1[rr][c0 + j]) =
          *reinterpret_cast<const uint4*>(qt + rowoff + j);
      *reinterpret_cast<uint4*>(&buf2[rr][c0 + j]) =
          *reinterpret_cast<const uint4*>(kt + rowoff + j);
    }
  }
  __syncthreads();
  v8f accST[2][4], accO[2][4];
  zero_acc(accST);
  zero_acc(accO);
  // scores^T(s,c) = sum_d K[s][d] * Q[c][d]
  mm128(&buf2[0][0], &buf1[0][0], LDC, wm, wn, lane, accST);
  __syncthreads();

  // phase 2: buf2 = KV^T[e][d] (straight copy), inter = rowscale(Q @ KV)
  {
    const __bf16* kvsrc = kvinT + (bh * NCz + nc) * (size_t)(Dz * Dz);
#pragma unroll
    for (int j = 0; j < 64; j += 8)
      *reinterpret_cast<uint4*>(&buf2[rr][c0 + j]) =
          *reinterpret_cast<const uint4*>(kvsrc + (size_t)rr * Dz + c0 + j);
  }
  __syncthreads();
  mm128(&buf1[0][0], &buf2[0][0], LDC, wm, wn, lane, accO);
#pragma unroll
  for (int i = 0; i < 2; ++i)
#pragma unroll
    for (int rg = 0; rg < 8; ++rg) {
      const float sc = exp2f(L * (float)(wm + i * 16 + rg + 8 * hi + 1));  // q_decay
#pragma unroll
      for (int j = 0; j < 4; ++j) accO[i][j][rg] *= sc;
    }
  __syncthreads();

  // phase 3: buf2 <- dmask(scores)[c][s] (packed v8bf), buf1 <- V^T[e][s]
#pragma unroll
  for (int i = 0; i < 2; ++i)
#pragma unroll
    for (int j = 0; j < 4; ++j) {
      const int c  = wn + j * 16 + nn;           // accST n-index
      const int sb = wm + i * 16 + 8 * hi;       // 8 consecutive s
      v8bf pk;
#pragma unroll
      for (int rg = 0; rg < 8; ++rg) {
        const int s = sb + rg;
        float v = accST[i][j][rg];
        v = (c >= s) ? v * exp2f(L * (float)(c - s)) : 0.f;
        pk[rg] = (__bf16)v;
      }
      *reinterpret_cast<v8bf*>(&buf2[c][sb]) = pk;
    }
  {
    const __bf16* vsrc = vtT + ((size_t)bh * Dz + rr) * Tz + t0 + c0;
#pragma unroll
    for (int j = 0; j < 64; j += 8)
      *reinterpret_cast<uint4*>(&buf1[rr][c0 + j]) =
          *reinterpret_cast<const uint4*>(vsrc + j);
  }
  __syncthreads();
  // intra(c,e) += sum_s S[c][s] * V[s][e]
  mm128(&buf2[0][0], &buf1[0][0], LDC, wm, wn, lane, accO);

  // store [c][h*128+e] tile to attn [B*T, E] (bf16, pre-RMS)
#pragma unroll
  for (int i = 0; i < 2; ++i)
#pragma unroll
    for (int j = 0; j < 4; ++j) {
      const int e = wn + j * 16 + nn;
#pragma unroll
      for (int rg = 0; rg < 8; ++rg) {
        const int c = wm + i * 16 + rg + 8 * hi;
        attnb[((size_t)b * Tz + (size_t)t0 + c) * Ez + h * Dz + e] =
            (__bf16)accO[i][j][rg];
      }
    }
}

// ------------------------------ RMSNorm (in place) ---------------------------

__global__ __launch_bounds__(256) void rms_kernel(__bf16* __restrict__ a,
                                                  const float* __restrict__ w) {
  __bf16* x = a + (size_t)blockIdx.x * Ez;
  const int tid = threadIdx.x, lane = tid & 31, wave = tid >> 5;
  float vals[8], local = 0.f;
#pragma unroll
  for (int i = 0; i < 8; ++i) {
    const float v = (float)x[tid + i * 256];
    vals[i] = v;
    local += v * v;
  }
#pragma unroll
  for (int off = 16; off > 0; off >>= 1) local += __shfl_xor(local, off, 32);
  __shared__ float red[8];
  if (lane == 0) red[wave] = local;
  __syncthreads();
  if (tid == 0) {
    float t = 0.f;
#pragma unroll
    for (int i = 0; i < 8; ++i) t += red[i];
    red[0] = rsqrtf(t * (1.0f / Ez) + 1e-5f);
  }
  __syncthreads();
  const float inv = red[0];
#pragma unroll
  for (int i = 0; i < 8; ++i) {
    const int cc = tid + i * 256;
    x[cc] = (__bf16)(vals[i] * inv * w[cc]);
  }
}

// ------------------------------- host driver ---------------------------------

extern "C" void kernel_launch(void* const* d_in, const int* in_sizes, int n_in,
                              void* d_out, int out_size, void* d_ws, size_t ws_size,
                              hipStream_t stream) {
  const float* x     = (const float*)d_in[0];
  const float* Wq    = (const float*)d_in[1];
  const float* Wk    = (const float*)d_in[2];
  const float* Wv    = (const float*)d_in[3];
  const float* Wo    = (const float*)d_in[4];
  const float* rms_w = (const float*)d_in[5];

  char* ws = (char*)d_ws;
  const size_t MB = 1ull << 20;
  __bf16* xb    = (__bf16*)(ws + 0);          // 64 MB (reused as attnb later)
  __bf16* wqbT  = (__bf16*)(ws + 64  * MB);   // [N,K] bf16, 8 MB each
  __bf16* wkbT  = (__bf16*)(ws + 72  * MB);
  __bf16* wvbT  = (__bf16*)(ws + 80  * MB);
  __bf16* wobT  = (__bf16*)(ws + 88  * MB);
  __bf16* qt    = (__bf16*)(ws + 96  * MB);   // [B,H,T,D] bf16
  __bf16* kt    = (__bf16*)(ws + 160 * MB);   // [B,H,T,D] bf16
  __bf16* vtT   = (__bf16*)(ws + 224 * MB);   // [B,H,D,T] bf16
  __bf16* ktT   = (__bf16*)(ws + 288 * MB);   // [B,H,D,T] bf16 (dead after chunk_kv)
  __bf16* kvcT  = (__bf16*)(ws + 352 * MB);   // [B,H,NC,D,D] (e-major) bf16
  __bf16* kvinT = ktT;                        // reuse ktT region (total 416 MB)
  __bf16* attnb = xb;                         // x dead after projections

  // 1. fp32 -> bf16 conversions (weights transposed to [N,K])
  cvt_f32_bf16<<<(Mz * Ez / 4 + 255) / 256, 256, 0, stream>>>(x, xb, Mz * Ez / 4);
  const dim3 tgrid(Ez / 32, Ez / 32);
  cvt_transpose_w<<<tgrid, 256, 0, stream>>>(Wq, wqbT);
  cvt_transpose_w<<<tgrid, 256, 0, stream>>>(Wk, wkbT);
  cvt_transpose_w<<<tgrid, 256, 0, stream>>>(Wv, wvbT);
  cvt_transpose_w<<<tgrid, 256, 0, stream>>>(Wo, wobT);

  // 2. QKV projections (WMMA GEMM; q,k -> [B,H,T,D]; v -> [B,H,D,T])
  const dim3 ggrid(Ez / 128, Mz / 128);
  gemm_bf16_128<1><<<ggrid, 256, 0, stream>>>(xb, wqbT, qt);
  gemm_bf16_128<1><<<ggrid, 256, 0, stream>>>(xb, wkbT, kt);
  gemm_bf16_128<2><<<ggrid, 256, 0, stream>>>(xb, wvbT, vtT);

  // 3. RoPE in place on q and k, then k -> k^T copy for chunk_kv
  const int rope_blocks = (Bz * Hz * Tz * 64) / 256;   // 65536
  rope_kernel<<<rope_blocks, 256, 0, stream>>>(qt);
  rope_kernel<<<rope_blocks, 256, 0, stream>>>(kt);
  transpose_dt<<<dim3(Tz / 32, Dz / 32, Bz * Hz), 256, 0, stream>>>(kt, ktT);

  // 4. lightning attention: per-chunk KV -> scan -> per-chunk output
  const dim3 cgrid(NCz, Hz, Bz);
  chunk_kv_kernel<<<cgrid, 256, 0, stream>>>(ktT, vtT, kvcT);
  kv_scan_kernel<<<(Bz * Hz * Dz * Dz) / 256, 256, 0, stream>>>(kvcT, kvinT);
  chunk_out_kernel<<<cgrid, 256, 0, stream>>>(qt, kt, vtT, kvinT, attnb);

  // 5. RMSNorm in place, then output projection (f32 out)
  rms_kernel<<<Mz, 256, 0, stream>>>(attnb, rms_w);
  gemm_bf16_128<0><<<ggrid, 256, 0, stream>>>(attnb, wobT, d_out);

  (void)in_sizes; (void)n_in; (void)out_size; (void)ws_size;
}
